// MambaAgents_73744588472362
// MI455X (gfx1250) — compile-verified, matrix operand order checked by reading, not run on hardware
//
#include <hip/hip_runtime.h>

// ============================================================================
// MambaAgents forward for MI455X (gfx1250, wave32).
// Big GEMMs -> v_wmma_f32_16x16x32_bf16 (bf16 operands, f32 accumulate),
// 32x64 C-tile per wave: 12 global_load_b128 -> 8 WMMAs per 32-K step.
// All GEMMs are NT (B pre-transposed by the cvt kernels); no strided gathers.
// Scan / softmax / layernorm / conv epilogues -> fp32 VALU + LDS.
// ============================================================================

#define L_SEQ   4096
#define NVIEW   4
#define D_INP   1024
#define DMODEL  512
#define DINNER  1024
#define DSTATE  16
#define DTRANK  32
#define NLAYERS 2
#define HEADS   8
#define DHEAD   64
#define LAND    256
#define LSEG    16

typedef __attribute__((ext_vector_type(16))) __bf16        bf16x16;
typedef __attribute__((ext_vector_type(8)))  float         f32x8;
typedef __attribute__((ext_vector_type(4)))  unsigned int  u32x4;

// explicit global-address-space views (force global_load/store, not flat)
typedef const __attribute__((address_space(1))) u32x4*  gc_u32x4p;
typedef __attribute__((address_space(1))) float*        g_f32p;
typedef const __attribute__((address_space(1))) float*  gc_f32p;

#define DEV static __device__ __forceinline__

DEV u32x4 gload16(const unsigned short* p) {          // 16B global load
  return *(gc_u32x4p)(unsigned long long)p;
}

DEV unsigned short f2bf(float f) {
  unsigned int u = __float_as_uint(f);
  return (unsigned short)((u + 0x7FFFu + ((u >> 16) & 1u)) >> 16);
}
DEV float sigm(float x) { return 1.0f / (1.0f + __expf(-x)); }

DEV float block_reduce_sum(float v, float* red) {
  int t = threadIdx.x;
  red[t] = v; __syncthreads();
  for (int o = 128; o > 0; o >>= 1) { if (t < o) red[t] += red[t + o]; __syncthreads(); }
  float r = red[0]; __syncthreads();
  return r;
}
DEV float block_reduce_max(float v, float* red) {
  int t = threadIdx.x;
  red[t] = v; __syncthreads();
  for (int o = 128; o > 0; o >>= 1) { if (t < o) red[t] = fmaxf(red[t], red[t + o]); __syncthreads(); }
  float r = red[0]; __syncthreads();
  return r;
}

// ---------------------------------------------------------------------------
// Batched NT WMMA GEMM.  C[b] = alpha * A[b] * B[b]^T (+bias)(act) + beta*C[b]
// A:[M,K], B:[N,K] row-major bf16 (ushort); C f32.
// M%32==0, N%64==0, K%32==0, lda/ldb multiples of 8, bases 16B-aligned.
// Block = 128 threads = 4 waves; each wave owns a 32(M) x 64(N) C tile.
// Per 32-K step: 6 fragments = 12 global_load_b128 -> 8 v_wmma_f32_16x16x32_bf16.
// ---------------------------------------------------------------------------
union FragAB { bf16x16 v; u32x4 q[2]; };

__global__ void gemm_wmma(const unsigned short* __restrict__ A, long long sA, int lda,
                          const unsigned short* __restrict__ B, long long sB, int ldb,
                          float* __restrict__ C, long long sC, int ldc,
                          int M, int N, int K,
                          const float* __restrict__ bias, int act, float alpha, float beta) {
  const int wave = threadIdx.x >> 5;
  const int lane = threadIdx.x & 31;
  const int r  = lane & 15;
  const int hi = lane >> 4;
  const int tn = blockIdx.x * 4 + wave;   // 64-wide N tile
  const int tm = blockIdx.y;              // 32-wide M tile
  if (tn * 64 >= N) return;               // wave-uniform exit

  A += (long long)blockIdx.z * sA;
  B += (long long)blockIdx.z * sB;
  C += (long long)blockIdx.z * sC;

  // documented 16-bit operand layout: lanes0-15 hold K{0..7,16..23},
  // lanes16-31 hold K{8..15,24..31} -> two contiguous 16B quads per fragment.
  const int qo1 = hi * 8;        // elements
  const int qo2 = 16 + hi * 8;

  int rA0 = tm * 32 + r;      if (rA0 >= M) rA0 = M - 1;
  int rA1 = tm * 32 + 16 + r; if (rA1 >= M) rA1 = M - 1;
  const unsigned short* pa[2] = { A + (long long)rA0 * lda, A + (long long)rA1 * lda };

  const unsigned short* pb[4];
#pragma unroll
  for (int j = 0; j < 4; ++j) {
    int rB = tn * 64 + j * 16 + r; if (rB >= N) rB = N - 1;
    pb[j] = B + (long long)rB * ldb;
  }

  f32x8 acc[2][4];
#pragma unroll
  for (int mi = 0; mi < 2; ++mi)
#pragma unroll
    for (int nj = 0; nj < 4; ++nj)
#pragma unroll
      for (int i = 0; i < 8; ++i) acc[mi][nj][i] = 0.0f;

  for (int kk = 0; kk < K; kk += 32) {
    FragAB a[2], b[4];
#pragma unroll
    for (int mi = 0; mi < 2; ++mi) {
      __builtin_prefetch(pa[mi] + 512, 0, 3);
      a[mi].q[0] = gload16(pa[mi] + qo1);
      a[mi].q[1] = gload16(pa[mi] + qo2);
    }
#pragma unroll
    for (int nj = 0; nj < 4; ++nj) {
      __builtin_prefetch(pb[nj] + 512, 0, 3);
      b[nj].q[0] = gload16(pb[nj] + qo1);
      b[nj].q[1] = gload16(pb[nj] + qo2);
    }
#pragma unroll
    for (int mi = 0; mi < 2; ++mi)
#pragma unroll
      for (int nj = 0; nj < 4; ++nj)
        acc[mi][nj] = __builtin_amdgcn_wmma_f32_16x16x32_bf16(
            false, a[mi].v, false, b[nj].v, (short)0, acc[mi][nj], false, false);
    pa[0] += 32; pa[1] += 32;
#pragma unroll
    for (int nj = 0; nj < 4; ++nj) pb[nj] += 32;
  }

#pragma unroll
  for (int mi = 0; mi < 2; ++mi)
#pragma unroll
    for (int nj = 0; nj < 4; ++nj)
#pragma unroll
      for (int i = 0; i < 8; ++i) {
        int gm = tm * 32 + mi * 16 + i + hi * 8;
        int gn = tn * 64 + nj * 16 + r;
        if (gm < M && gn < N) {
          float vv = alpha * acc[mi][nj][i] + (bias ? ((gc_f32p)(unsigned long long)(bias + gn))[0] : 0.0f);
          if (act == 1)      vv = fmaxf(vv, 0.0f);                        // relu
          else if (act == 2) vv = (vv > 20.0f) ? vv : log1pf(__expf(vv)); // softplus
          g_f32p cp = (g_f32p)(unsigned long long)(C + (long long)gm * ldc + gn);
          *cp = (beta != 0.0f ? beta * (*cp) : 0.0f) + vv;
        }
      }
}

// ---------------------------------------------------------------------------
// Conversions
// ---------------------------------------------------------------------------
__global__ void cvt_bf16_1d(const float* __restrict__ in, unsigned short* __restrict__ out, long long n) {
  long long i = (long long)blockIdx.x * 256 + threadIdx.x;
  if (i < n) out[i] = f2bf(in[i]);
}
__global__ void cvt_bf16_2d(const float* __restrict__ in, int ldin,
                            unsigned short* __restrict__ out, int ldout, int cols) {
  int row = blockIdx.x;
  for (int c = threadIdx.x; c < cols; c += blockDim.x)
    out[(long long)row * ldout + c] = f2bf(in[(long long)row * ldin + c]);
}
// batched transpose-convert: in [b,rows,cols] f32 -> out [b,cols,rows] bf16
__global__ void cvt_t_bf16(const float* __restrict__ in, unsigned short* __restrict__ out,
                           int rows, int cols, long long n) {
  long long idx = (long long)blockIdx.x * 256 + threadIdx.x;
  if (idx >= n) return;
  long long rc = (long long)rows * cols;
  long long b = idx / rc; long long rem = idx - b * rc;
  int rr = (int)(rem / cols), cc = (int)(rem % cols);
  out[b * rc + (long long)cc * rows + rr] = f2bf(in[idx]);
}
// v^T extraction: qkv [L,1536] v-cols -> vt [H,64,L] bf16
__global__ void vt_extract(const float* __restrict__ qkvf, unsigned short* __restrict__ vt) {
  long long idx = (long long)blockIdx.x * 256 + threadIdx.x; // h*262144 + d*4096 + l
  int l = (int)(idx & 4095); int d = (int)((idx >> 12) & 63); int hh = (int)(idx >> 18);
  vt[idx] = f2bf(qkvf[(long long)l * 1536 + 1024 + hh * 64 + d]);
}

// ---------------------------------------------------------------------------
// LayerNorm over D, writes bf16 rows (GEMM A operand)
// ---------------------------------------------------------------------------
__global__ void layernorm_bf16(const float* __restrict__ x, long long rowStride,
                               const float* __restrict__ g, const float* __restrict__ b,
                               unsigned short* __restrict__ y, int D) {
  __shared__ float red[256];
  int row = blockIdx.x;
  const float* xr = x + (long long)row * rowStride;
  float s = 0.f;
  for (int d = threadIdx.x; d < D; d += 256) s += xr[d];
  float mu = block_reduce_sum(s, red) / D;
  float v = 0.f;
  for (int d = threadIdx.x; d < D; d += 256) { float t = xr[d] - mu; v += t * t; }
  float inv = rsqrtf(block_reduce_sum(v, red) / D + 1e-5f);
  for (int d = threadIdx.x; d < D; d += 256)
    y[(long long)row * D + d] = f2bf((xr[d] - mu) * inv * g[d] + b[d]);
}

// causal depthwise conv (k=4) + SiLU; reads xc half of xz [L,2048]
__global__ void conv1d_silu(const float* __restrict__ xz, const float* __restrict__ w,
                            const float* __restrict__ bias, float* __restrict__ of,
                            unsigned short* __restrict__ ob) {
  long long idx = (long long)blockIdx.x * 256 + threadIdx.x; // l*1024+d
  int d = (int)(idx & 1023); long long l = idx >> 10;
  float acc = bias[d];
#pragma unroll
  for (int k = 0; k < 4; ++k) {
    long long ll = l + k - 3;
    if (ll >= 0) acc += xz[ll * 2048 + d] * w[d * 4 + k];
  }
  float s = acc * sigm(acc);
  of[idx] = s; ob[idx] = f2bf(s);
}

// selective scan: 1024 channels (4 blocks x 256), sequential over L.
// epilogue: y = (scan + x*D) * silu(z) -> bf16
__global__ void mamba_scan(const float* __restrict__ dtf, const float* __restrict__ xcf,
                           const float* __restrict__ xdbl, const float* __restrict__ xz,
                           const float* __restrict__ A_log, const float* __restrict__ Dp,
                           unsigned short* __restrict__ yb, int L) {
  int d = blockIdx.x * 256 + threadIdx.x;
  float As[DSTATE], hh[DSTATE];
#pragma unroll
  for (int s = 0; s < DSTATE; ++s) { As[s] = -__expf(A_log[d * DSTATE + s]); hh[s] = 0.f; }
  float Dd = Dp[d];
  __shared__ float BC[32];
  for (int l = 0; l < L; ++l) {
    if (threadIdx.x < 32) BC[threadIdx.x] = xdbl[(long long)l * 64 + 32 + threadIdx.x];
    __syncthreads();
    float dt  = dtf[(long long)l * DINNER + d];
    float xv  = xcf[(long long)l * DINNER + d];
    float dtx = dt * xv;
    float y = 0.f;
#pragma unroll
    for (int s = 0; s < DSTATE; ++s) {
      float dA = __expf(dt * As[s]);
      hh[s] = dA * hh[s] + dtx * BC[s];
      y += hh[s] * BC[16 + s];
    }
    float zv  = xz[(long long)l * 2048 + 1024 + d];
    float out = (y + xv * Dd) * (zv * sigm(zv));
    yb[(long long)l * DINNER + d] = f2bf(out);
    __syncthreads();
  }
}

// h[l,v,:] += mean_v' mvout[l,v',:]
__global__ void mv_merge(float* __restrict__ h, const float* __restrict__ mv) {
  long long idx = (long long)blockIdx.x * 256 + threadIdx.x; // l*512+d
  long long l = idx >> 9; int d = (int)(idx & 511);
  long long base = l * 2048 + d;
  float m = 0.25f * (mv[base] + mv[base + 512] + mv[base + 1024] + mv[base + 1536]);
  h[base] += m; h[base + 512] += m; h[base + 1024] += m; h[base + 1536] += m;
}
__global__ void mean_views(const float* __restrict__ h, float* __restrict__ hs) {
  long long idx = (long long)blockIdx.x * 256 + threadIdx.x;
  long long l = idx >> 9; int d = (int)(idx & 511);
  long long base = l * 2048 + d;
  hs[idx] = 0.25f * (h[base] + h[base + 512] + h[base + 1024] + h[base + 1536]);
}

// landmark means (q unscaled; 1/sqrt(dh) folded into GEMM alpha)
__global__ void landmarks_k(const float* __restrict__ qkvf,
                            unsigned short* __restrict__ qlb, unsigned short* __restrict__ klb) {
  int idx = blockIdx.x * 256 + threadIdx.x;          // h*16384 + i*64 + d
  int d = idx & 63; int i = (idx >> 6) & 255; int hh = idx >> 14;
  int col = hh * 64 + d;
  float sq = 0.f, sk = 0.f;
  for (int j = 0; j < LSEG; ++j) {
    long long row = (long long)(i * LSEG + j) * 1536;
    sq += qkvf[row + col];
    sk += qkvf[row + 512 + col];
  }
  qlb[idx] = f2bf(sq * (1.0f / LSEG));
  klb[idx] = f2bf(sk * (1.0f / LSEG));
}

__global__ void softmax_rows(float* __restrict__ x, int cols) {
  __shared__ float red[256];
  float* xr = x + (long long)blockIdx.x * cols;
  float mx = -1e30f;
  for (int c = threadIdx.x; c < cols; c += 256) mx = fmaxf(mx, xr[c]);
  mx = block_reduce_max(mx, red);
  float s = 0.f;
  for (int c = threadIdx.x; c < cols; c += 256) { float e = __expf(xr[c] - mx); xr[c] = e; s += e; }
  float inv = 1.0f / block_reduce_sum(s, red);
  for (int c = threadIdx.x; c < cols; c += 256) xr[c] *= inv;
}

// pinv helpers (batched [8,256,256])
__global__ void abs_rowsum(const float* __restrict__ x, float* __restrict__ out, int cols) {
  __shared__ float red[256];
  const float* xr = x + (long long)blockIdx.x * cols;
  float s = 0.f;
  for (int c = threadIdx.x; c < cols; c += 256) s += fabsf(xr[c]);
  float t = block_reduce_sum(s, red);
  if (threadIdx.x == 0) out[blockIdx.x] = t;
}
__global__ void abs_colsum(const float* __restrict__ x, float* __restrict__ out, int rows, int cols) {
  int idx = blockIdx.x * 256 + threadIdx.x;          // b*cols + j
  int b = idx >> 8, j = idx & 255;
  const float* xb = x + (long long)b * rows * cols;
  float s = 0.f;
  for (int i = 0; i < rows; ++i) s += fabsf(xb[(long long)i * cols + j]);
  out[idx] = s;
}
__global__ void pinv_scale(const float* __restrict__ colb, const float* __restrict__ rowb,
                           float* __restrict__ scal, int n) {
  __shared__ float red[256];
  float mc = -1e30f, mr = -1e30f;
  for (int i = threadIdx.x; i < n; i += 256) { mc = fmaxf(mc, colb[i]); mr = fmaxf(mr, rowb[i]); }
  float M1 = block_reduce_max(mc, red);
  float M2 = block_reduce_max(mr, red);
  if (threadIdx.x == 0) scal[0] = 1.0f / (M1 * M2);
}
__global__ void transpose_scale(const float* __restrict__ x, float* __restrict__ z,
                                const float* __restrict__ scal) {
  long long idx = (long long)blockIdx.x * 256 + threadIdx.x; // b*65536 + i*256 + j
  int j = (int)(idx & 255); int i = (int)((idx >> 8) & 255); long long b = idx >> 16;
  z[idx] = x[(b << 16) + (long long)j * 256 + i] * scal[0];
}
// t^T = (c*I - x)^T written transposed (output is only ever a B operand)
__global__ void diag_sub_t(const float* __restrict__ x, float* __restrict__ t, float c) {
  long long idx = (long long)blockIdx.x * 256 + threadIdx.x; // b*65536 + i*256 + j
  int j = (int)(idx & 255); int i = (int)((idx >> 8) & 255); long long b = idx >> 16;
  t[(b << 16) + (long long)j * 256 + i] = (i == j ? c : 0.0f) - x[idx];
}

// residual depthwise head conv (33 taps) on v, added into out[h,l,d]
__global__ void resconv_add(const float* __restrict__ qkvf, const float* __restrict__ rw,
                            float* __restrict__ w3, int L) {
  long long idx = (long long)blockIdx.x * 256 + threadIdx.x; // h*L*64 + l*64 + d
  int d = (int)(idx & 63); long long l = (idx >> 6) & (L - 1); int hh = (int)(idx >> 18);
  int col = 1024 + hh * 64 + d;
  float acc = 0.f;
  for (int j = 0; j < 33; ++j) {
    long long ll = l + j - 16;
    if (ll >= 0 && ll < L) acc += qkvf[ll * 1536 + col] * rw[hh * 33 + j];
  }
  w3[idx] += acc;
}
__global__ void merge_heads(const float* __restrict__ w3, unsigned short* __restrict__ out) {
  long long idx = (long long)blockIdx.x * 256 + threadIdx.x; // l*512 + h*64 + d
  int d = (int)(idx & 63); int hh = (int)((idx >> 6) & 7); long long l = idx >> 9;
  out[idx] = f2bf(w3[((long long)hh << 18) + l * 64 + d]);
}

// final reductions
__global__ void feat_mean(const float* __restrict__ hs, float* __restrict__ out) {
  int d = blockIdx.x * 256 + threadIdx.x;
  float s = 0.f;
  for (int l = 0; l < L_SEQ; ++l) s += hs[(long long)l * DMODEL + d];
  out[d] = s * (1.0f / L_SEQ);
}
__global__ void logits_k(const float* __restrict__ feat, const float* __restrict__ cw,
                         const float* __restrict__ cb, float* __restrict__ out) {
  __shared__ float red[256];
  int c = blockIdx.x;
  float s = 0.f;
  for (int d = threadIdx.x; d < DMODEL; d += 256) s += feat[d] * cw[c * DMODEL + d];
  float t = block_reduce_sum(s, red);
  if (threadIdx.x == 0) out[c] = t + cb[c];
}
__global__ void af_mean(const float* __restrict__ p2, float* __restrict__ out) {
  int idx = threadIdx.x;                 // v*64 + c
  int c = idx & 63, v = idx >> 6;
  float s = 0.f;
  for (int l = 0; l < L_SEQ; ++l) s += p2[((long long)l * NVIEW + v) * 64 + c];
  out[idx] = s * (1.0f / L_SEQ);
}

// ============================================================================
// Host orchestration
// ============================================================================
static void gemm(hipStream_t st, const unsigned short* A, long long sA, int lda,
                 const unsigned short* B, long long sB, int ldb,
                 float* C, long long sC, int ldc, int M, int N, int K,
                 const float* bias, int act, float alpha, float beta, int batch) {
  dim3 g((N + 255) / 256, (M + 31) / 32, batch);
  gemm_wmma<<<g, dim3(128), 0, st>>>(A, sA, lda, B, sB, ldb, C, sC, ldc,
                                     M, N, K, bias, act, alpha, beta);
}
static void cvt(hipStream_t st, const float* in, unsigned short* out, long long n) {
  cvt_bf16_1d<<<dim3((unsigned)((n + 255) / 256)), dim3(256), 0, st>>>(in, out, n);
}
static void cvtT(hipStream_t st, const float* in, unsigned short* out,
                 int rows, int cols, long long n) {
  cvt_t_bf16<<<dim3((unsigned)((n + 255) / 256)), dim3(256), 0, st>>>(in, out, rows, cols, n);
}

// bf16 weight pool offsets (ushort units)
enum : long long {
  WB_INPROJ = 0,
  WB_INW    = WB_INPROJ + 524288LL,     // [2,4,2048,512]
  WB_XPROJ  = WB_INW    + 8388608LL,    // [2,4,64,1024]
  WB_DTW    = WB_XPROJ  + 524288LL,     // [2,4,1024,32]
  WB_MOUT   = WB_DTW    + 262144LL,     // [2,4,512,1024]
  WB_QKV    = WB_MOUT   + 4194304LL,    // [2,1536,512]
  WB_TOUT   = WB_QKV    + 1572864LL,    // [2,512,512]
  WB_PROJ1  = WB_TOUT   + 524288LL,     // [512,512]
  WB_PROJ2  = WB_PROJ1  + 262144LL      // [64,512]
};

// workspace byte offsets (phase-aliased)
static const size_t OFF_H    = 33554432;               // f32 [L,V,512]
static const size_t OFF_MV   = OFF_H    + 33554432;    // f32 [L,V,512]  (agent)
static const size_t OFF_S0   = OFF_MV   + 33554432;    // x_bf | xz f32 | sim1/a1 f32 | proj2-out
static const size_t OFF_S0B  = OFF_S0   + 33554432;    // ybf | a1 bf16
static const size_t OFF_S1   = OFF_S0B  + 16777216;    // convx_f+dt_f | sim3/a3 f32 | proj1-out
static const size_t OFF_S1B  = OFF_S1   + 33554432;    // convx_b+xdbl+dtlow | a3 bf16 | proj1-out bf16
static const size_t OFF_QKVF = OFF_S1B  + 16777216;    // f32 [L,1536]
static const size_t OFF_QKVB = OFF_QKVF + 25165824;    // bf16 [L,1536]
static const size_t OFF_ABF  = OFF_QKVB + 12582912;    // bf16 [16384,512] (LN out / agent bf)
static const size_t OFF_POOL = OFF_ABF  + 16777216;
// pool sub-offsets (bytes)
static const size_t PO_QLB = 0,        PO_KLB = 262144;
static const size_t PO_A2F = 524288,   PO_A2B = 2621440;
static const size_t PO_ZF  = 3670016,  PO_ZB  = 5767168;
static const size_t PO_XZF = 6815744,  PO_XZB = 8912896;
static const size_t PO_TF  = 9961472,  PO_TB  = 12058624;
static const size_t PO_UF  = 13107200;
static const size_t PO_W1F = 15204352, PO_W1B = 15728640;
static const size_t PO_W2F = 15990784, PO_W2B = 16515072;
static const size_t PO_W3F = 16777216;                 // f32 [8,4096,64]
static const size_t PO_ATTNB = 25165824;               // bf16 [4096,512]
static const size_t PO_HSEQ  = 29360128;               // f32 [4096,512]
static const size_t PO_COLB  = 37748736, PO_ROWB = 37756928, PO_SCAL = 37765120;
static const size_t PO_VTB = 37765632;                 // bf16 [H,64,4096] v^T
static const size_t PO_ZBT = PO_VTB + 4194304;         // bf16 z^T [8,256,256]
// w1t/w2t (bf16 [8,64,256]) alias the (free) W1B/W2B slots

struct Params {
  const float *x, *in_proj_w, *in_proj_b;
  const float *ln_g, *ln_b, *inW, *convW, *convB, *xprojW, *dtW, *dtB, *A_log, *Dm, *m_outW;
  const float *norm_g, *norm_b, *qkvW, *t_outW, *outB, *resW;
  const float *cls_w, *cls_b, *proj1, *proj2;
};

extern "C" void kernel_launch(void* const* d_in, const int* in_sizes, int n_in,
                              void* d_out, int out_size, void* d_ws, size_t ws_size,
                              hipStream_t stream) {
  Params P;
  auto F = [&](int i) { return (const float*)d_in[i]; };
  if (n_in >= 24 && in_sizes[0] == 2) {
    // jax tree_leaves (alphabetical) order: params first (cls_b=2 leads)
    P.cls_b = F(0);  P.cls_w = F(1);  P.in_proj_b = F(2);  P.in_proj_w = F(3);
    P.A_log = F(4);  P.Dm = F(5);     P.convB = F(6);      P.convW = F(7);
    P.dtB = F(8);    P.dtW = F(9);    P.inW = F(10);       P.ln_b = F(11);
    P.ln_g = F(12);  P.m_outW = F(13); P.xprojW = F(14);
    P.proj1 = F(15); P.proj2 = F(16);
    P.norm_b = F(17); P.norm_g = F(18); P.outB = F(19); P.t_outW = F(20);
    P.qkvW = F(21);  P.resW = F(22);  P.x = F(23);
  } else {
    // setup_inputs() insertion order
    P.x = F(0); P.in_proj_w = F(1); P.in_proj_b = F(2);
    P.ln_g = F(3); P.ln_b = F(4); P.inW = F(5); P.convW = F(6); P.convB = F(7);
    P.xprojW = F(8); P.dtW = F(9); P.dtB = F(10); P.A_log = F(11); P.Dm = F(12); P.m_outW = F(13);
    P.norm_g = F(14); P.norm_b = F(15); P.qkvW = F(16); P.t_outW = F(17); P.outB = F(18); P.resW = F(19);
    P.cls_w = F(20); P.cls_b = F(21); P.proj1 = F(22); P.proj2 = F(23);
  }

  char* W = (char*)d_ws;
  auto FP = [&](size_t off) { return (float*)(W + off); };
  auto UP = [&](size_t off) { return (unsigned short*)(W + off); };
  unsigned short* WB = UP(0);
  float* out = (float*)d_out;

  // ---- convert weights + input to bf16 ----
  cvt(stream, P.in_proj_w, WB + WB_INPROJ, 524288);
  cvt(stream, P.inW,       WB + WB_INW,    8388608);
  cvt(stream, P.xprojW,    WB + WB_XPROJ,  524288);
  cvt(stream, P.dtW,       WB + WB_DTW,    262144);
  cvt(stream, P.m_outW,    WB + WB_MOUT,   4194304);
  cvt(stream, P.qkvW,      WB + WB_QKV,    1572864);
  cvt(stream, P.t_outW,    WB + WB_TOUT,   524288);
  cvt(stream, P.proj1,     WB + WB_PROJ1,  262144);
  cvt(stream, P.proj2,     WB + WB_PROJ2,  32768);
  cvt(stream, P.x,         UP(OFF_S0),     16777216);

  // ---- in_proj + relu : h[L*V,512] ----
  gemm(stream, UP(OFF_S0), 0, D_INP, WB + WB_INPROJ, 0, D_INP,
       FP(OFF_H), 0, DMODEL, L_SEQ * NVIEW, DMODEL, D_INP, P.in_proj_b, 1, 1.f, 0.f, 1);

  // ---- mamba layers ----
  float*          xzf   = FP(OFF_S0);
  float*          cvf   = FP(OFF_S1);
  float*          dtf   = FP(OFF_S1 + 16777216);
  unsigned short* cvb   = UP(OFF_S1B);
  float*          xdbl  = FP(OFF_S1B + 8388608);
  unsigned short* dtlow = UP(OFF_S1B + 9437184);
  unsigned short* ybf   = UP(OFF_S0B);

  for (int i = 0; i < NLAYERS; ++i) {
    for (int v = 0; v < NVIEW; ++v) {
      long long iv = (long long)i * NVIEW + v;
      layernorm_bf16<<<L_SEQ, 256, 0, stream>>>(FP(OFF_H) + v * DMODEL, (long long)NVIEW * DMODEL,
                                                P.ln_g + iv * DMODEL, P.ln_b + iv * DMODEL,
                                                UP(OFF_ABF), DMODEL);
      gemm(stream, UP(OFF_ABF), 0, DMODEL, WB + WB_INW + iv * 2 * DINNER * DMODEL, 0, DMODEL,
           xzf, 0, 2 * DINNER, L_SEQ, 2 * DINNER, DMODEL, nullptr, 0, 1.f, 0.f, 1);
      conv1d_silu<<<16384, 256, 0, stream>>>(xzf, P.convW + iv * DINNER * 4,
                                             P.convB + iv * DINNER, cvf, cvb);
      gemm(stream, cvb, 0, DINNER, WB + WB_XPROJ + iv * 64 * DINNER, 0, DINNER,
           xdbl, 0, 64, L_SEQ, 64, DINNER, nullptr, 0, 1.f, 0.f, 1);
      cvt_bf16_2d<<<L_SEQ, 256, 0, stream>>>(xdbl, 64, dtlow, DTRANK, DTRANK);
      gemm(stream, dtlow, 0, DTRANK, WB + WB_DTW + iv * DINNER * DTRANK, 0, DTRANK,
           dtf, 0, DINNER, L_SEQ, DINNER, DTRANK, P.dtB + iv * DINNER, 2, 1.f, 0.f, 1);
      mamba_scan<<<4, 256, 0, stream>>>(dtf, cvf, xdbl, xzf,
                                        P.A_log + iv * DINNER * DSTATE, P.Dm + iv * DINNER,
                                        ybf, L_SEQ);
      gemm(stream, ybf, 0, DINNER, WB + WB_MOUT + iv * DMODEL * DINNER, 0, DINNER,
           FP(OFF_MV) + v * DMODEL, 0, NVIEW * DMODEL, L_SEQ, DMODEL, DINNER,
           nullptr, 0, 1.f, 0.f, 1);
    }
    mv_merge<<<8192, 256, 0, stream>>>(FP(OFF_H), FP(OFF_MV)); // OFF_MV of last layer == agent
  }

  float* hseq = FP(OFF_POOL + PO_HSEQ);
  mean_views<<<8192, 256, 0, stream>>>(FP(OFF_H), hseq);

  // ---- transformer (Nystrom) layers ----
  float*          qkvf = FP(OFF_QKVF);
  unsigned short* qkvb = UP(OFF_QKVB);
  unsigned short* qlb  = UP(OFF_POOL + PO_QLB);
  unsigned short* klb  = UP(OFF_POOL + PO_KLB);
  float* a2f = FP(OFF_POOL + PO_A2F);  unsigned short* a2b = UP(OFF_POOL + PO_A2B);
  float* zf  = FP(OFF_POOL + PO_ZF);   unsigned short* zb  = UP(OFF_POOL + PO_ZB);
  float* xzp = FP(OFF_POOL + PO_XZF);  unsigned short* xzb = UP(OFF_POOL + PO_XZB);
  float* tf  = FP(OFF_POOL + PO_TF);   unsigned short* tb  = UP(OFF_POOL + PO_TB);
  float* uf  = FP(OFF_POOL + PO_UF);
  float* w1f = FP(OFF_POOL + PO_W1F);  unsigned short* w1t = UP(OFF_POOL + PO_W1B);
  float* w2f = FP(OFF_POOL + PO_W2F);  unsigned short* w2t = UP(OFF_POOL + PO_W2B);
  float* w3f = FP(OFF_POOL + PO_W3F);
  unsigned short* attnb = UP(OFF_POOL + PO_ATTNB);
  unsigned short* vtb   = UP(OFF_POOL + PO_VTB);
  unsigned short* zbt   = UP(OFF_POOL + PO_ZBT);
  float* s0f = FP(OFF_S0);  unsigned short* s0b = UP(OFF_S0B);
  float* s1f = FP(OFF_S1);  unsigned short* s1b = UP(OFF_S1B);
  float* colb = FP(OFF_POOL + PO_COLB);
  float* rowb = FP(OFF_POOL + PO_ROWB);
  float* scal = FP(OFF_POOL + PO_SCAL);
  const float sc = 0.125f; // DIM_HEAD^-0.5

  for (int i = 0; i < NLAYERS; ++i) {
    layernorm_bf16<<<L_SEQ, 256, 0, stream>>>(hseq, DMODEL, P.norm_g + i * DMODEL,
                                              P.norm_b + i * DMODEL, UP(OFF_ABF), DMODEL);
    gemm(stream, UP(OFF_ABF), 0, DMODEL, WB + WB_QKV + (long long)i * 1536 * DMODEL, 0, DMODEL,
         qkvf, 0, 1536, L_SEQ, 1536, DMODEL, nullptr, 0, 1.f, 0.f, 1);
    cvt(stream, qkvf, qkvb, (long long)L_SEQ * 1536);
    vt_extract<<<8192, 256, 0, stream>>>(qkvf, vtb);
    landmarks_k<<<512, 256, 0, stream>>>(qkvf, qlb, klb);
    // sim1 [H,L,256], sim2 [H,256,256], sim3 [H,256,L]
    gemm(stream, qkvb, 64, 1536, klb, 16384, 64, s0f, 1048576, 256,
         L_SEQ, LAND, DHEAD, nullptr, 0, sc, 0.f, HEADS);
    gemm(stream, qlb, 16384, 64, klb, 16384, 64, a2f, 65536, 256,
         LAND, LAND, DHEAD, nullptr, 0, sc, 0.f, HEADS);
    gemm(stream, qlb, 16384, 64, qkvb + 512, 64, 1536, s1f, 1048576, 4096,
         LAND, L_SEQ, DHEAD, nullptr, 0, sc, 0.f, HEADS);
    softmax_rows<<<HEADS * L_SEQ, 256, 0, stream>>>(s0f, LAND);
    softmax_rows<<<HEADS * LAND, 256, 0, stream>>>(a2f, LAND);
    softmax_rows<<<HEADS * LAND, 256, 0, stream>>>(s1f, L_SEQ);
    cvt(stream, s0f, s0b, 8388608);
    cvt(stream, s1f, s1b, 8388608);
    cvt(stream, a2f, a2b, 524288);
    // Moore-Penrose pinv of a2
    abs_rowsum<<<HEADS * LAND, 256, 0, stream>>>(a2f, colb, LAND);
    abs_colsum<<<8, 256, 0, stream>>>(a2f, rowb, LAND, LAND);
    pinv_scale<<<1, 256, 0, stream>>>(colb, rowb, scal, HEADS * LAND);
    transpose_scale<<<2048, 256, 0, stream>>>(a2f, zf, scal);
    for (int it = 0; it < 6; ++it) {
      cvt(stream, zf, zb, 524288);                       // z as A operand
      cvtT(stream, zf, zbt, LAND, LAND, 524288);         // z^T as B operand
      gemm(stream, a2b, 65536, 256, zbt, 65536, 256, xzp, 65536, 256,
           LAND, LAND, LAND, nullptr, 0, 1.f, 0.f, HEADS);
      cvt(stream, xzp, xzb, 524288);
      diag_sub_t<<<2048, 256, 0, stream>>>(xzp, tf, 7.f);
      cvt(stream, tf, tb, 524288);
      gemm(stream, xzb, 65536, 256, tb, 65536, 256, uf, 65536, 256,
           LAND, LAND, LAND, nullptr, 0, 1.f, 0.f, HEADS);
      diag_sub_t<<<2048, 256, 0, stream>>>(uf, tf, 15.f);
      cvt(stream, tf, tb, 524288);
      gemm(stream, xzb, 65536, 256, tb, 65536, 256, uf, 65536, 256,
           LAND, LAND, LAND, nullptr, 0, 1.f, 0.f, HEADS);
      diag_sub_t<<<2048, 256, 0, stream>>>(uf, tf, 13.f);
      cvt(stream, tf, tb, 524288);
      gemm(stream, zb, 65536, 256, tb, 65536, 256, zf, 65536, 256,
           LAND, LAND, LAND, nullptr, 0, 0.25f, 0.f, HEADS);
    }
    cvt(stream, zf, zb, 524288);   // a2inv (bf16, A operand)
    // out = a1 @ (a2inv @ (a3 @ v))
    gemm(stream, s1b, 1048576, 4096, vtb, 262144, 4096, w1f, 16384, 64,
         LAND, DHEAD, L_SEQ, nullptr, 0, 1.f, 0.f, HEADS);
    cvtT(stream, w1f, w1t, LAND, DHEAD, 131072);
    gemm(stream, zb, 65536, 256, w1t, 16384, 256, w2f, 16384, 64,
         LAND, DHEAD, LAND, nullptr, 0, 1.f, 0.f, HEADS);
    cvtT(stream, w2f, w2t, LAND, DHEAD, 131072);
    gemm(stream, s0b, 1048576, 256, w2t, 16384, 256, w3f, 262144, 64,
         L_SEQ, DHEAD, LAND, nullptr, 0, 1.f, 0.f, HEADS);
    resconv_add<<<8192, 256, 0, stream>>>(qkvf, P.resW + (long long)i * HEADS * 33, w3f, L_SEQ);
    merge_heads<<<8192, 256, 0, stream>>>(w3f, attnb);
    // residual: hseq = hseq + attn @ outW^T + outB
    gemm(stream, attnb, 0, DMODEL, WB + WB_TOUT + (long long)i * DMODEL * DMODEL, 0, DMODEL,
         hseq, 0, DMODEL, L_SEQ, DMODEL, DMODEL, P.outB + i * DMODEL, 0, 1.f, 1.f, 1);
  }

  // ---- outputs: features, logits, h, af ----
  feat_mean<<<2, 256, 0, stream>>>(hseq, out);
  logits_k<<<2, 256, 0, stream>>>(out, P.cls_w, P.cls_b, out + 512);
  hipMemcpyAsync(out + 514, hseq, (size_t)L_SEQ * DMODEL * sizeof(float),
                 hipMemcpyDeviceToDevice, stream);
  // af = mean_l( relu(agent @ proj1^T) @ proj2^T )
  cvt(stream, FP(OFF_MV), UP(OFF_ABF), 8388608);
  gemm(stream, UP(OFF_ABF), 0, DMODEL, WB + WB_PROJ1, 0, DMODEL,
       FP(OFF_S1), 0, DMODEL, L_SEQ * NVIEW, DMODEL, DMODEL, nullptr, 1, 1.f, 0.f, 1);
  cvt(stream, FP(OFF_S1), UP(OFF_S1B), 8388608);
  gemm(stream, UP(OFF_S1B), 0, DMODEL, WB + WB_PROJ2, 0, DMODEL,
       FP(OFF_S0), 0, 64, L_SEQ * NVIEW, 64, DMODEL, nullptr, 0, 1.f, 0.f, 1);
  af_mean<<<1, 256, 0, stream>>>(FP(OFF_S0), out + 514 + L_SEQ * DMODEL);
  (void)n_in; (void)out_size; (void)ws_size;
}
